// MHMoEHPNRT_4621384810815
// MI455X (gfx1250) — matrix operand order repaired and checked by reading, not compiled, not run on hardware
//
#include <hip/hip_runtime.h>
#include <hip/hip_bf16.h>

// ---------------------------------------------------------------------------
// MH-MoE forward for gfx1250 (MI455X).
// - Everything converted to bf16 once (64MB K/V streams from HBM exactly once;
//   bf16 working set then lives in the 192MB L2 for expert-block reuse).
// - All matmuls via v_wmma_f32_16x16x32_bf16. GEMMs use a 32x64 wave tile with
//   fixed-role ping-pong fragment buffers (no copy chains, loads overlap MMAs).
//   GEMM epilogue is templated on output type so q is written as bf16 directly.
// - Block-sparse expert stage batches selections per (head,expert); K is
//   staged to LDS via ASYNCcnt-tracked global_load_async_to_lds_b128 (when the
//   builtin exists), V is staged transposed, so every B fragment is two
//   contiguous b128 loads.
// ---------------------------------------------------------------------------

typedef __attribute__((ext_vector_type(16))) __bf16 v16bf;
typedef __attribute__((ext_vector_type(8)))  float  v8f;

#define BATH 512     // B*T
#define E_   1024
#define H_   8
#define HS_  128
#define NE_  64
#define ES_  128
#define CAP  512     // max tokens per (head,expert): top-2 distinct -> <= BATH
#define KL_LD 136    // padded LDS leading dims (17*16B rows: aligned, bank-spread)
#define VT_LD 136
#define PL_LD 136

#if __has_builtin(__builtin_amdgcn_global_load_async_to_lds_b128)
#define HAVE_ASYNC_LDS 1
// Exact parameter types per hipcc diagnostic: v4i in AS1 (global) / AS3 (LDS).
typedef __attribute__((__vector_size__(16))) int v4i_;
typedef __attribute__((address_space(1))) v4i_ g_v4i;
typedef __attribute__((address_space(3))) v4i_ l_v4i;
#endif

__device__ __forceinline__ unsigned short f2bf(float f) {
  unsigned int u = __float_as_uint(f);
  u += 0x7FFFu + ((u >> 16) & 1u);   // round-to-nearest-even
  return (unsigned short)(u >> 16);
}

template <typename T>
__device__ __forceinline__ void store_out(T* p, float v);
template <>
__device__ __forceinline__ void store_out<float>(float* p, float v) { *p = v; }
template <>
__device__ __forceinline__ void store_out<unsigned short>(unsigned short* p, float v) {
  *p = f2bf(v);
}

// A fragment (16x32 bf16): caller pre-offsets p by (hi?16:0);
// lane holds 16 consecutive K: p[0..15].
__device__ __forceinline__ v16bf load_frag_a(const unsigned short* p) {
  union { v16bf v; uint4 q[2]; } f;
  f.q[0] = *(const uint4*)(p);
  f.q[1] = *(const uint4*)(p + 8);
  return f.v;
}

// B fragment (32x16 bf16), NT source: lane holds 16 consecutive K of column n;
// caller pre-offsets by (hi?16:0).
__device__ __forceinline__ v16bf load_frag_b_cont(const unsigned short* p) {
  union { v16bf v; uint4 q[2]; } f;
  f.q[0] = *(const uint4*)(p);
  f.q[1] = *(const uint4*)(p + 8);
  return f.v;
}

__device__ __forceinline__ v8f wmma_bf16(v16bf a, v16bf b, v8f c) {
  return __builtin_amdgcn_wmma_f32_16x16x32_bf16(false, a, false, b, (short)0, c,
                                                 false, false);
}

// ---------------------------------------------------------------------------
// bf16 conversion, 4 floats / thread (pure bandwidth pass; n must be %4)
__global__ void cvt_kernel(const float* __restrict__ src,
                           unsigned short* __restrict__ dst, int n4) {
  int i = blockIdx.x * blockDim.x + threadIdx.x;
  if (i < n4) {
    float4 v = ((const float4*)src)[i];
    union { unsigned short u[4]; uint2 q; } o;
    o.u[0] = f2bf(v.x); o.u[1] = f2bf(v.y);
    o.u[2] = f2bf(v.z); o.u[3] = f2bf(v.w);
    ((uint2*)dst)[i] = o.q;
  }
}

// router [H][HS][NE] f32 -> router_t [H][NE][HS] bf16 (so scores GEMM is NT)
__global__ void cvt_router_t_kernel(const float* __restrict__ r,
                                    unsigned short* __restrict__ rt) {
  int h = blockIdx.x;
  for (int i = threadIdx.x; i < HS_ * NE_; i += 256) {
    int d = i >> 6, e = i & 63;                      // coalesced read over e
    rt[(size_t)h * NE_ * HS_ + (size_t)e * HS_ + d] =
        f2bf(r[(size_t)h * HS_ * NE_ + i]);
  }
}

__global__ void zero_i32_kernel(int* __restrict__ p, int n) {
  int i = blockIdx.x * blockDim.x + threadIdx.x;
  if (i < n) p[i] = 0;
}

// vsum[h][e][d] = sum_s V[h][e][s][d]   (reversal-trick offset, f32 source)
__global__ void vsum_kernel(const float* __restrict__ v, float* __restrict__ vsum) {
  int he = blockIdx.x;
  int d  = threadIdx.x;
  const float* p = v + (size_t)he * ES_ * HS_ + d;
  float s = 0.f;
  for (int ss = 0; ss < ES_; ++ss) s += p[(size_t)ss * HS_];
  vsum[(size_t)he * HS_ + d] = s;
}

// ---------------------------------------------------------------------------
// C[M,N] = A[M,K](bf16 row-major) * B[N,K](bf16 row-major over K) ("NT")
// One wave: 32x64 tile (2 A-frags x 4 B-frags -> 8 WMMAs per 12 b128 loads).
// Two fixed-role fragment buffer sets (ping/pong), K%64==0 required.
// Output type templated: f32, or bf16 (fused downconvert epilogue).
template <typename OutT>
__global__ void __launch_bounds__(256) gemm_nt_kernel(
    const unsigned short* __restrict__ A, const unsigned short* __restrict__ B,
    OutT* __restrict__ C, int M, int N, int K, int lda, int ldb, int ldc,
    int nh, int aH, int bH, int cH) {
  int wave = threadIdx.x >> 5, lane = threadIdx.x & 31;
  bool hiv = lane >= 16; int nl = lane & 15;
  int tiles_m = M >> 5, tiles_n = N >> 6;
  int per_h = tiles_m * tiles_n;
  int tid = blockIdx.x * 8 + wave;
  if (tid >= nh * per_h) return;                   // wave-uniform
  int hh = tid / per_h; int r0 = tid - hh * per_h;
  int tm = r0 / tiles_n, tn = r0 - tm * tiles_n;
  int row0 = tm << 5, col0 = tn << 6;
  const unsigned short* ap =
      A + (size_t)hh * aH + (size_t)(row0 + nl) * lda + (hiv ? 16 : 0);
  size_t astep = (size_t)16 * lda;
  const unsigned short* bp =
      B + (size_t)hh * bH + (size_t)(col0 + nl) * ldb + (hiv ? 16 : 0);
  size_t bstep = (size_t)16 * ldb;

  v16bf aP[2], bP[4], aQ[2], bQ[4];
  v8f acc[2][4] = {};

  auto load_buf = [&](v16bf (&af)[2], v16bf (&bf)[4], int kk) {
    af[0] = load_frag_a(ap + kk);
    af[1] = load_frag_a(ap + astep + kk);
#pragma unroll
    for (int j = 0; j < 4; ++j) bf[j] = load_frag_b_cont(bp + j * bstep + kk);
  };
  auto mma_buf = [&](v16bf (&af)[2], v16bf (&bf)[4]) {
#pragma unroll
    for (int i = 0; i < 2; ++i)
#pragma unroll
      for (int j = 0; j < 4; ++j) acc[i][j] = wmma_bf16(af[i], bf[j], acc[i][j]);
  };

  load_buf(aP, bP, 0);
  load_buf(aQ, bQ, 32);
  for (int kk = 64; kk < K; kk += 64) {
    mma_buf(aP, bP);
    load_buf(aP, bP, kk);
    mma_buf(aQ, bQ);
    load_buf(aQ, bQ, kk + 32);
  }
  mma_buf(aP, bP);
  mma_buf(aQ, bQ);

  OutT* Cp = C + (size_t)hh * cH;
#pragma unroll
  for (int i = 0; i < 2; ++i)
#pragma unroll
    for (int j = 0; j < 4; ++j)
#pragma unroll
      for (int r = 0; r < 8; ++r)
        store_out(Cp + (size_t)(row0 + 16 * i + r + (hiv ? 8 : 0)) * ldc +
                      col0 + 16 * j + nl,
                  acc[i][j][r]);
}

// ---------------------------------------------------------------------------
// Top-2 on biased scores, softmax gates on raw scores, per-(h,e) bucketing.
// Each selection owns a unique o-slot -> deterministic output regardless of
// atomic ordering.
__global__ void topk_kernel(const float* __restrict__ scores,
                            const float* __restrict__ bias,
                            int* __restrict__ idx, float* __restrict__ gate,
                            int* __restrict__ cnt, int* __restrict__ list) {
  int t = blockIdx.x * blockDim.x + threadIdx.x;   // h*BATH + b
  if (t >= H_ * BATH) return;
  int h = t >> 9;
  const float* s  = scores + (size_t)t * NE_;
  const float* bi = bias + h * NE_;
  int i1 = 0, i2 = 0;
  float v1 = -3.4e38f, v2 = -3.4e38f;
  for (int e = 0; e < NE_; ++e) {
    float f = s[e] + bi[e];
    if (f > v1)      { v2 = v1; i2 = i1; v1 = f; i1 = e; }
    else if (f > v2) { v2 = f; i2 = e; }
  }
  float r1 = s[i1], r2 = s[i2];
  float mx = fmaxf(r1, r2);
  float e1 = __expf(r1 - mx), e2 = __expf(r2 - mx);
  float den = e1 + e2;
  idx[t * 2 + 0] = i1; idx[t * 2 + 1] = i2;
  gate[t * 2 + 0] = e1 / den; gate[t * 2 + 1] = e2 / den;
  int b2 = (t & 511) * 2;
  int slot = atomicAdd(&cnt[h * NE_ + i1], 1);
  list[(size_t)(h * NE_ + i1) * CAP + slot] = b2;
  slot = atomicAdd(&cnt[h * NE_ + i2], 1);
  list[(size_t)(h * NE_ + i2) * CAP + slot] = b2 + 1;
}

// ---------------------------------------------------------------------------
// Per-(head,expert): stage K into LDS (ASYNCcnt-tracked async-to-LDS copies
// when available), stage V transposed, then for each 16-token chunk:
//   logits = Qc[16x128] * K_e^T      (WMMA, contiguous LDS B frags)
//   P = exp(logits) -> LDS (re-layout C->A fragment)
//   O = P * V_e                       (WMMA, contiguous LDS B frags from Vt)
//   o = O - Vsum, scatter per selection.
__global__ void __launch_bounds__(256) moe_attn_kernel(
    const unsigned short* __restrict__ qb,   // [BATH][E] bf16
    const unsigned short* __restrict__ kb,   // [H][NE][ES][HS] bf16
    const unsigned short* __restrict__ vb,   // [H][NE][ES][HS] bf16
    const float* __restrict__ vsum,          // [H][NE][HS]
    const int* __restrict__ cnt, const int* __restrict__ list,
    float* __restrict__ o_out) {             // [H][BATH][2][HS]
  extern __shared__ __align__(16) unsigned short smem[];
  int he = blockIdx.x;
  int h = he >> 6;
  int nt = cnt[he];
  if (nt == 0) return;
  int wave = threadIdx.x >> 5, lane = threadIdx.x & 31;
  bool hiv = lane >= 16; int nl = lane & 15;
  const unsigned short* Km = kb + (size_t)he * ES_ * HS_;
  const unsigned short* Vm = vb + (size_t)he * ES_ * HS_;
  const float* vs = vsum + (size_t)he * HS_;
  const int* lst = list + (size_t)he * CAP;
  unsigned short* Kl = smem;                       // [ES][KL_LD]  (s-major)
  unsigned short* Vt = Kl + ES_ * KL_LD;           // [HS][VT_LD]  (d-major)
  unsigned short* pl = Vt + HS_ * VT_LD;           // [8][16*PL_LD]

  // ---- stage K into LDS (async DMA path if the toolchain exposes it) ----
#ifdef HAVE_ASYNC_LDS
  {
    g_v4i* gk = (g_v4i*)Km;          // 16B units; global row stride = 16 units
    l_v4i* lk = (l_v4i*)Kl;          // padded LDS row stride = 272B = 17 units
    for (int cid = threadIdx.x; cid < (ES_ * HS_) / 8; cid += 256) {
      int s = cid >> 4, cc = cid & 15;             // 16B chunk cc of row s
      __builtin_amdgcn_global_load_async_to_lds_b128(gk + s * 16 + cc,
                                                     lk + s * 17 + cc, 0, 0);
    }
#if __has_builtin(__builtin_amdgcn_s_wait_asynccnt)
    __builtin_amdgcn_s_wait_asynccnt(0);
#else
    asm volatile("s_wait_asynccnt 0x0" ::: "memory");
#endif
  }
#else
  for (int i = threadIdx.x; i < ES_ * HS_; i += 256) {
    int s = i >> 7, d = i & 127;
    Kl[s * KL_LD + d] = Km[i];
  }
#endif
  // ---- stage V transposed (coalesced global reads; padded LDS stride) ----
  for (int i = threadIdx.x; i < ES_ * HS_; i += 256) {
    int s = i >> 7, d = i & 127;
    Vt[d * VT_LD + s] = Vm[i];
  }
  __syncthreads();

  unsigned short* pw = pl + wave * 16 * PL_LD;
  int nchunk = (nt + 15) >> 4;
  for (int c = wave; c < nchunk; c += 8) {         // wave-uniform loop
    int base = c << 4;
    int rem = nt - base; if (rem > 16) rem = 16;
    int tokA = lst[base + (nl < rem ? nl : 0)];    // padded rows read token 0
    const unsigned short* qrow =
        qb + (size_t)(tokA >> 1) * E_ + h * HS_ + (hiv ? 16 : 0);

    // logits = Qc * K^T : M=16, N=128 (s), K=128 (d)
    v8f accL[8] = {};
    for (int kk = 0; kk < HS_; kk += 32) {
      v16bf a = load_frag_a(qrow + kk);
      v16bf b[8];
#pragma unroll
      for (int j = 0; j < 8; ++j)
        b[j] = load_frag_b_cont(Kl + (size_t)(16 * j + nl) * KL_LD + kk +
                                (hiv ? 16 : 0));
#pragma unroll
      for (int j = 0; j < 8; ++j) accL[j] = wmma_bf16(a, b[j], accL[j]);
    }
    // exp() -> LDS, re-layout C(row=m,col=s) to [m][s]
#pragma unroll
    for (int j = 0; j < 8; ++j) {
      int s = 16 * j + nl;
#pragma unroll
      for (int r = 0; r < 8; ++r)
        pw[(r + (hiv ? 8 : 0)) * PL_LD + s] = f2bf(__expf(accL[j][r]));
    }
    // O = P * V : M=16, N=128 (d), K=128 (s); B frags contiguous from Vt
    v8f accO[8] = {};
    for (int kk = 0; kk < ES_; kk += 32) {
      v16bf a = load_frag_a(pw + nl * PL_LD + kk + (hiv ? 16 : 0));
      v16bf b[8];
#pragma unroll
      for (int j = 0; j < 8; ++j)
        b[j] = load_frag_b_cont(Vt + (size_t)(16 * j + nl) * VT_LD + kk +
                                (hiv ? 16 : 0));
#pragma unroll
      for (int j = 0; j < 8; ++j) accO[j] = wmma_bf16(a, b[j], accO[j]);
    }
    // epilogue: o = O - Vsum, scatter per selection
#pragma unroll
    for (int r = 0; r < 8; ++r) {
      int m = r + (hiv ? 8 : 0);
      if (m < rem) {
        int tok = lst[base + m];                   // b*2 + k
        float* orow = o_out + ((size_t)h * (BATH * 2) + tok) * HS_;
#pragma unroll
        for (int j = 0; j < 8; ++j) {
          int d = 16 * j + nl;
          orow[d] = accO[j][r] - vs[d];
        }
      }
    }
  }
}

// agg[b][h*HS+d] (bf16) = sum_k gate[h][b][k] * o[h][b][k][d]
__global__ void agg_kernel(const float* __restrict__ o,
                           const float* __restrict__ gate,
                           unsigned short* __restrict__ aggb) {
  int t = blockIdx.x;             // h*BATH + b
  int d = threadIdx.x;            // 0..127
  int h = t >> 9, b = t & 511;
  float g0 = gate[t * 2 + 0], g1 = gate[t * 2 + 1];
  const float* ob = o + (size_t)t * 2 * HS_;
  float v = g0 * ob[d] + g1 * ob[HS_ + d];
  aggb[(size_t)b * E_ + h * HS_ + d] = f2bf(v);
}

// ---------------------------------------------------------------------------
extern "C" void kernel_launch(void* const* d_in, const int* in_sizes, int n_in,
                              void* d_out, int out_size, void* d_ws, size_t ws_size,
                              hipStream_t stream) {
  const float* x      = (const float*)d_in[0];   // [2,256,1024]
  const float* w_mh   = (const float*)d_in[1];   // [1024,1024]
  const float* w_mg   = (const float*)d_in[2];   // [1024,1024]
  const float* router = (const float*)d_in[3];   // [8,128,64]
  const float* k_ffwd = (const float*)d_in[4];   // [8,8192,128]
  const float* v_ffwd = (const float*)d_in[5];   // [8,8192,128]
  const float* bias   = (const float*)d_in[6];   // [8,64]
  (void)in_sizes; (void)n_in; (void)out_size; (void)ws_size;

  char* ws = (char*)d_ws;
  size_t off = 0;
  auto take = [&](size_t bytes) -> char* {
    char* p = ws + off;
    off += (bytes + 255) & ~(size_t)255;
    return p;
  };
  unsigned short* xb   = (unsigned short*)take((size_t)BATH * E_ * 2);
  unsigned short* mhb  = (unsigned short*)take((size_t)E_ * E_ * 2);
  unsigned short* mgb  = (unsigned short*)take((size_t)E_ * E_ * 2);
  unsigned short* rtb  = (unsigned short*)take((size_t)H_ * NE_ * HS_ * 2); // transposed
  unsigned short* kbf  = (unsigned short*)take((size_t)H_ * NE_ * ES_ * HS_ * 2);
  unsigned short* vbf  = (unsigned short*)take((size_t)H_ * NE_ * ES_ * HS_ * 2);
  unsigned short* qbf  = (unsigned short*)take((size_t)BATH * E_ * 2);
  float*          sc   = (float*)take((size_t)H_ * BATH * NE_ * 4);
  int*            idx  = (int*)take((size_t)H_ * BATH * 2 * 4);
  float*          gate = (float*)take((size_t)H_ * BATH * 2 * 4);
  int*            cnt  = (int*)take((size_t)H_ * NE_ * 4);
  int*            list = (int*)take((size_t)H_ * NE_ * CAP * 4);
  float*          vsm  = (float*)take((size_t)H_ * NE_ * HS_ * 4);
  float*          ofp  = (float*)take((size_t)H_ * BATH * 2 * HS_ * 4);
  unsigned short* aggb = (unsigned short*)take((size_t)BATH * E_ * 2);

  auto cvt = [&](const float* s, unsigned short* d, int n) {
    cvt_kernel<<<(n / 4 + 255) / 256, 256, 0, stream>>>(s, d, n / 4);
  };
  // one-shot bf16 conversion: streams each f32 tensor exactly once from HBM
  cvt(x, xb, BATH * E_);
  cvt(w_mh, mhb, E_ * E_);
  cvt(w_mg, mgb, E_ * E_);
  cvt_router_t_kernel<<<H_, 256, 0, stream>>>(router, rtb);
  cvt(k_ffwd, kbf, H_ * NE_ * ES_ * HS_);
  cvt(v_ffwd, vbf, H_ * NE_ * ES_ * HS_);

  vsum_kernel<<<H_ * NE_, HS_, 0, stream>>>(v_ffwd, vsm);
  zero_i32_kernel<<<2, 256, 0, stream>>>(cnt, H_ * NE_);

  // q = x @ w_mh^T (M=512,N=1024,K=1024), fused bf16 epilogue -> qbf directly
  gemm_nt_kernel<unsigned short><<<32, 256, 0, stream>>>(
      xb, mhb, qbf, BATH, E_, E_, E_, E_, E_, 1, 0, 0, 0);

  // scores[h] = q[h] @ router_t[h]^T (M=512,N=64,K=128): 8*16=128 waves
  gemm_nt_kernel<float><<<16, 256, 0, stream>>>(
      qbf, rtb, sc, BATH, NE_, HS_, E_, HS_, NE_,
      H_, HS_, NE_ * HS_, BATH * NE_);

  topk_kernel<<<(H_ * BATH) / 256, 256, 0, stream>>>(sc, bias, idx, gate, cnt, list);

  size_t smem_bytes = (size_t)(ES_ * KL_LD + HS_ * VT_LD + 8 * 16 * PL_LD) * 2;
  moe_attn_kernel<<<H_ * NE_, 256, smem_bytes, stream>>>(qbf, kbf, vbf, vsm,
                                                         cnt, list, ofp);

  agg_kernel<<<H_ * BATH, HS_, 0, stream>>>(ofp, gate, aggb);

  // y = agg @ w_mg^T -> d_out (f32 [2,256,1024] == [512][1024])
  gemm_nt_kernel<float><<<32, 256, 0, stream>>>(
      aggb, mgb, (float*)d_out, BATH, E_, E_, E_, E_, E_, 1, 0, 0, 0);
}
